// CNNSCNNEncoder_7481833030315
// MI455X (gfx1250) — matrix operand order repaired and checked by reading, not compiled, not run on hardware
//
#include <hip/hip_runtime.h>
#include <hip/hip_bf16.h>

// ---------------------------------------------------------------------------
// CNN+SCNN encoder for MI455X (gfx1250, wave32).
// Implicit-GEMM convs on v_wmma_f32_16x16x32_f16.
//  - NHWC f16 activations; K reordered tap-major so each lane's 16 B-halves
//    are 16 contiguous channels -> one 32B vector load per lane per K-chunk.
//  - Weights pre-packed f16 in WMMA A-fragment order (one 32B load / chunk).
//  - Branchless, clamped-address gathers: loads are unconditional so the
//    backend batches them into clauses (no serialized load+wait).
//  - Deterministic two-stage BN statistics (no float atomics).
// ---------------------------------------------------------------------------

typedef __attribute__((ext_vector_type(16))) _Float16 v16h;
typedef __attribute__((ext_vector_type(8)))  float    v8f;

__device__ __forceinline__ int iclamp(int v, int lo, int hi) {
  return v < lo ? lo : (v > hi ? hi : v);
}

// ---- x (4,3,288,800) NCHW f32 -> NHWC f16 -----------------------------------
__global__ __launch_bounds__(256)
void cvt_in_kernel(const float* __restrict__ x, _Float16* __restrict__ out) {
  size_t i = (size_t)blockIdx.x * blockDim.x + threadIdx.x;
  const size_t tot = (size_t)4 * 3 * 288 * 800;
  if (i >= tot) return;
  int c = (int)(i % 3);
  size_t p = i / 3;
  int xw = (int)(p % 800);
  size_t t = p / 800;
  int y = (int)(t % 288);
  int n = (int)(t / 288);
  out[i] = (_Float16)x[((size_t)(n * 3 + c) * 288 + y) * 800 + xw];
}

// ---- repack f32 weights (Co x Ci x KH x KW) into f16 WMMA A-fragments -------
// GEMM K is tap-major: kk = tap*Ci + ci. frag[((ct*kChunks+kc)*32+lane)*16+j]
__global__ __launch_bounds__(256)
void repack_w_kernel(const float* __restrict__ w, _Float16* __restrict__ out,
                     int Co, int Ci, int khkw, int coTiles, int kChunks) {
  size_t i = (size_t)blockIdx.x * blockDim.x + threadIdx.x;
  size_t tot = (size_t)coTiles * kChunks * 512;
  if (i >= tot) return;
  int j  = (int)(i & 15);
  int l  = (int)((i >> 4) & 31);
  size_t c = i >> 9;
  int kc = (int)(c % (size_t)kChunks);
  int ct = (int)(c / (size_t)kChunks);
  int n16 = l & 15, hi = l >> 4;
  int Ktot = Ci * khkw;
  int kk = kc * 32 + hi * 16 + j;
  int co = ct * 16 + n16;
  float v = 0.f;
  if (kk < Ktot && co < Co) {
    int tap = kk / Ci, ci = kk - tap * Ci;
    v = w[(size_t)co * Ktot + (size_t)ci * khkw + tap];
  }
  out[i] = (_Float16)v;
}

// ---- implicit-GEMM conv2d, NHWC f16 in / NHWC f32 out, Ci % 16 == 0 ---------
template<int KH, int KW, int CI, int NT>
__global__ __launch_bounds__(256)
void conv_wmma_vec_kernel(const _Float16* __restrict__ in,
                          const _Float16* __restrict__ wfrag,
                          const float* __restrict__ bias, float* __restrict__ out,
                          int H, int W, int Co, int pad, int dil,
                          int coTiles, int grpTiles) {
  const int khkw = KH * KW;
  const int Ktot = CI * khkw;
  const int kChunks = (Ktot + 31) / 32;
  int waveId = blockIdx.x * (blockDim.x >> 5) + (threadIdx.x >> 5);
  if (waveId >= coTiles * grpTiles) return;
  int grp    = waveId / coTiles;
  int coTile = waveId - grp * coTiles;
  int lane = threadIdx.x & 31;
  int n16 = lane & 15, hi = lane >> 4;
  int P = 4 * H * W;

  int ox[NT], oy[NT], img[NT]; bool pv[NT];
#pragma unroll
  for (int t = 0; t < NT; ++t) {
    int p = (grp * NT + t) * 16 + n16;
    pv[t] = p < P;
    int pc = pv[t] ? p : 0;
    ox[t] = pc % W; int tq = pc / W; oy[t] = tq % H; img[t] = tq / H;
  }

  v8f zero = {};
  v16h zeroh = {};
  v8f acc[NT];
#pragma unroll
  for (int t = 0; t < NT; ++t) acc[t] = zero;

  const _Float16* wbase = wfrag + ((size_t)coTile * kChunks * 32 + lane) * 16;
  for (int kc = 0; kc < kChunks; ++kc) {
    v16h a = *(const v16h*)(wbase + (size_t)kc * 512);
    int kbase = kc * 32 + hi * 16;           // lane-half's 16 K's: one tap, 16 ci
    bool kvalid = kbase < Ktot;
    int tap = kvalid ? kbase / CI : 0;       // CI power-of-2 -> shift
    int cibase = kbase - tap * CI;
    int ky = tap / KW, kx = tap - ky * KW;
    int dy = ky * dil - pad, dx = kx * dil - pad;
#pragma unroll
    for (int t = 0; t < NT; ++t) {
      int iy = oy[t] + dy, ix = ox[t] + dx;
      bool ok = pv[t] && kvalid && iy >= 0 && iy < H && ix >= 0 && ix < W;
      int iyc = iclamp(iy, 0, H - 1), ixc = iclamp(ix, 0, W - 1);
      v16h b = *(const v16h*)(in + (((size_t)img[t] * H + iyc) * W + ixc) * CI + cibase);
      if (!ok) b = zeroh;
      acc[t] = __builtin_amdgcn_wmma_f32_16x16x32_f16(false, a, false, b,
                                                      (short)0, acc[t],
                                                      false, false);
    }
    // speculative prefetch of next chunk's activation line (global_prefetch_b8)
    if (kc + 1 < kChunks) {
      int kb2 = kbase + 32;
      int tap2 = kb2 < Ktot ? kb2 / CI : 0;
      int cib2 = kb2 - tap2 * CI;
      int ky2 = tap2 / KW, kx2 = tap2 - ky2 * KW;
      int iy2 = iclamp(oy[0] + ky2 * dil - pad, 0, H - 1);
      int ix2 = iclamp(ox[0] + kx2 * dil - pad, 0, W - 1);
      __builtin_prefetch(in + (((size_t)img[0] * H + iy2) * W + ix2) * CI + cib2, 0, 3);
    }
  }
  // NHWC epilogue: 8 contiguous co per lane -> merged wide stores
#pragma unroll
  for (int t = 0; t < NT; ++t) {
    if (!pv[t]) continue;
    int co0 = coTile * 16 + hi * 8;
    float* po = out + (((size_t)img[t] * H + oy[t]) * W + ox[t]) * Co + co0;
#pragma unroll
    for (int r = 0; r < 8; ++r)
      po[r] = acc[t][r] + (bias ? bias[co0 + r] : 0.f);
  }
}

// ---- general (scalar-gather) conv for Ci=3 first layer ----------------------
template<int KH, int KW, int CI, int NT>
__global__ __launch_bounds__(256)
void conv_wmma_gen_kernel(const _Float16* __restrict__ in,
                          const _Float16* __restrict__ wfrag,
                          const float* __restrict__ bias, float* __restrict__ out,
                          int H, int W, int Co, int pad, int dil,
                          int coTiles, int grpTiles) {
  const int khkw = KH * KW;
  const int Ktot = CI * khkw;
  const int kChunks = (Ktot + 31) / 32;
  int waveId = blockIdx.x * (blockDim.x >> 5) + (threadIdx.x >> 5);
  if (waveId >= coTiles * grpTiles) return;
  int grp    = waveId / coTiles;
  int coTile = waveId - grp * coTiles;
  int lane = threadIdx.x & 31;
  int n16 = lane & 15, hi = lane >> 4;
  int P = 4 * H * W;

  int ox[NT], oy[NT], img[NT]; bool pv[NT];
#pragma unroll
  for (int t = 0; t < NT; ++t) {
    int p = (grp * NT + t) * 16 + n16;
    pv[t] = p < P;
    int pc = pv[t] ? p : 0;
    ox[t] = pc % W; int tq = pc / W; oy[t] = tq % H; img[t] = tq / H;
  }
  v8f zero = {};
  v8f acc[NT];
#pragma unroll
  for (int t = 0; t < NT; ++t) acc[t] = zero;

  const _Float16* wbase = wfrag + ((size_t)coTile * kChunks * 32 + lane) * 16;
  for (int kc = 0; kc < kChunks; ++kc) {
    v16h a = *(const v16h*)(wbase + (size_t)kc * 512);
    int kbase = kc * 32 + hi * 16;
#pragma unroll
    for (int t = 0; t < NT; ++t) {
      v16h b;
#pragma unroll
      for (int j = 0; j < 16; ++j) {
        int kk = kbase + j;
        bool kv = kk < Ktot;
        int kkc = kv ? kk : 0;
        int tap = kkc / CI, ci = kkc - tap * CI;   // constant divisors
        int ky = tap / KW, kx = tap - ky * KW;
        int iy = oy[t] + ky * dil - pad;
        int ix = ox[t] + kx * dil - pad;
        bool ok = pv[t] && kv && iy >= 0 && iy < H && ix >= 0 && ix < W;
        int iyc = iclamp(iy, 0, H - 1), ixc = iclamp(ix, 0, W - 1);
        _Float16 hv = in[(((size_t)img[t] * H + iyc) * W + ixc) * CI + ci];
        b[j] = ok ? hv : (_Float16)0.f;
      }
      acc[t] = __builtin_amdgcn_wmma_f32_16x16x32_f16(false, a, false, b,
                                                      (short)0, acc[t],
                                                      false, false);
    }
  }
#pragma unroll
  for (int t = 0; t < NT; ++t) {
    if (!pv[t]) continue;
    int co0 = coTile * 16 + hi * 8;
    float* po = out + (((size_t)img[t] * H + oy[t]) * W + ox[t]) * Co + co0;
#pragma unroll
    for (int r = 0; r < 8; ++r)
      po[r] = acc[t][r] + (bias ? bias[co0 + r] : 0.f);
  }
}

// ---- BN stats, stage 1: contiguous sweep, thread t owns channel t%C ---------
// Requires C | 256. part[b*2C + c] = sum, part[b*2C + C + c] = sumsq.
__global__ __launch_bounds__(256)
void stats1_kernel(const float* __restrict__ raw, float* __restrict__ part,
                   int C, size_t total) {
  size_t stride = (size_t)gridDim.x * 256;
  size_t per = ((total + stride - 1) / stride) * 256;
  size_t beg = (size_t)blockIdx.x * per;
  size_t end = beg + per; if (end > total) end = total;
  float s = 0.f, q = 0.f;
  for (size_t e = beg + threadIdx.x; e < end; e += 256) {
    float v = raw[e]; s += v; q += v * v;
  }
  __shared__ float ss[256], sq[256];
  ss[threadIdx.x] = s; sq[threadIdx.x] = q;
  __syncthreads();
  for (int o = 128; o >= C; o >>= 1) {
    if ((int)threadIdx.x < o) {
      ss[threadIdx.x] += ss[threadIdx.x + o];
      sq[threadIdx.x] += sq[threadIdx.x + o];
    }
    __syncthreads();
  }
  if ((int)threadIdx.x < C) {
    part[(size_t)blockIdx.x * 2 * C + threadIdx.x]     = ss[threadIdx.x];
    part[(size_t)blockIdx.x * 2 * C + C + threadIdx.x] = sq[threadIdx.x];
  }
}

// ---- BN stats, stage 2: fixed-order sum over block partials (deterministic) -
__global__ __launch_bounds__(256)
void stats2_kernel(const float* __restrict__ part, float* __restrict__ st,
                   int C, int B) {
  int c = threadIdx.x;
  if (c >= C) return;
  float s = 0.f, q = 0.f;
  for (int b = 0; b < B; ++b) {
    s += part[(size_t)b * 2 * C + c];
    q += part[(size_t)b * 2 * C + C + c];
  }
  st[c] = s; st[C + c] = q;
}

// ---- BN + ReLU, NHWC raw f32 -> NHWC f16 act --------------------------------
__global__ __launch_bounds__(256)
void bn_relu_f16_kernel(const float* __restrict__ raw, const float* __restrict__ st,
                        const float* __restrict__ g, const float* __restrict__ bt,
                        _Float16* __restrict__ out, int C, size_t total, float cnt) {
  size_t i = (size_t)blockIdx.x * blockDim.x + threadIdx.x;
  if (i >= total) return;
  int c = (int)(i & (size_t)(C - 1));           // C is a power of two
  float m = st[c] / cnt;
  float v = st[C + c] / cnt - m * m;
  float y = (raw[i] - m) * rsqrtf(v + 1e-5f) * g[c] + bt[c];
  out[i] = (_Float16)(y > 0.f ? y : 0.f);
}

// ---- BN + ReLU, NHWC raw -> NCHW f32 (final h before SCNN) ------------------
__global__ __launch_bounds__(256)
void bn_relu_nchw_kernel(const float* __restrict__ raw, const float* __restrict__ st,
                         const float* __restrict__ g, const float* __restrict__ bt,
                         float* __restrict__ out, int C, int cShift,
                         int Hh, int Wh, size_t total, float cnt) {
  size_t i = (size_t)blockIdx.x * blockDim.x + threadIdx.x;
  if (i >= total) return;
  int c = (int)(i & (size_t)(C - 1));
  size_t p = i >> cShift;
  int xw = (int)(p % (size_t)Wh);
  size_t t = p / (size_t)Wh;
  int y = (int)(t % (size_t)Hh);
  int n = (int)(t / (size_t)Hh);
  float m = st[c] / cnt;
  float v = st[C + c] / cnt - m * m;
  float yv = (raw[i] - m) * rsqrtf(v + 1e-5f) * g[c] + bt[c];
  out[((size_t)(n * C + c) * Hh + y) * Wh + xw] = yv > 0.f ? yv : 0.f;
}

// ---- fused BN + ReLU + 2x2 maxpool: NHWC raw -> NHWC f16 act + NCHW f32 feat -
__global__ __launch_bounds__(256)
void bn_relu_pool_kernel(const float* __restrict__ raw, const float* __restrict__ st,
                         const float* __restrict__ g, const float* __restrict__ bt,
                         _Float16* __restrict__ outH, float* __restrict__ feat,
                         int C, int cShift, int H, int W, float cnt) {
  int Ho = H >> 1, Wo = W >> 1;
  size_t i = (size_t)blockIdx.x * blockDim.x + threadIdx.x;
  size_t tot = ((size_t)4 * Ho * Wo) << cShift;
  if (i >= tot) return;
  int c = (int)(i & (size_t)(C - 1));
  size_t p = i >> cShift;
  int xw = (int)(p % (size_t)Wo);
  size_t t = p / (size_t)Wo;
  int y = (int)(t % (size_t)Ho);
  int n = (int)(t / (size_t)Ho);
  float m = st[c] / cnt;
  float v = st[C + c] / cnt - m * m;
  float scale = rsqrtf(v + 1e-5f) * g[c];
  float shift = bt[c] - m * scale;
  const float* pp = raw + (((size_t)(n * H + 2 * y)) * W + 2 * xw) * C + c;
  size_t sC = (size_t)C, sWC = (size_t)W * C;
  float a0 = pp[0] * scale + shift;        a0 = a0 > 0.f ? a0 : 0.f;
  float a1 = pp[sC] * scale + shift;       a1 = a1 > 0.f ? a1 : 0.f;
  float a2 = pp[sWC] * scale + shift;      a2 = a2 > 0.f ? a2 : 0.f;
  float a3 = pp[sWC + sC] * scale + shift; a3 = a3 > 0.f ? a3 : 0.f;
  float mx = fmaxf(fmaxf(a0, a1), fmaxf(a2, a3));
  outH[i] = (_Float16)mx;
  if (feat) feat[((size_t)(n * C + c) * Ho + y) * Wo + xw] = mx;
}

// ---- one SCNN slice: h[dst] += relu(conv1d(h[src])), WMMA GEMM --------------
// h is NCHW f32 (4,128,18,50); kernel fragments tap-major (kk = tap*128 + ci).
__global__ __launch_bounds__(256)
void mp_step_wmma_kernel(float* __restrict__ h, const _Float16* __restrict__ kfrag,
                         int axisVert, int srcS, int dstS, int pixTiles) {
  const int C = 128, Hh = 18, Wh = 50, HW = Hh * Wh, kChunks = 36, coTiles = 8;
  int waveId = blockIdx.x * (blockDim.x >> 5) + (threadIdx.x >> 5);
  if (waveId >= coTiles * pixTiles) return;
  int pixTile = waveId / coTiles;
  int coTile  = waveId - pixTile * coTiles;
  int lane = threadIdx.x & 31;
  int n16 = lane & 15, hi = lane >> 4;
  int L    = axisVert ? Wh : Hh;
  int qMul = axisVert ? 1 : Wh;
  int sOfs = axisVert ? srcS * Wh : srcS;
  int P = 4 * L;
  int p = pixTile * 16 + n16;
  bool pvalid = p < P;
  int pc = pvalid ? p : 0;
  int pos = pc % L, img = pc / L;

  v8f acc = {};
  const _Float16* kbaseP = kfrag + ((size_t)coTile * kChunks * 32 + lane) * 16;
  for (int kc = 0; kc < kChunks; ++kc) {
    v16h a = *(const v16h*)(kbaseP + (size_t)kc * 512);
    int kbase = kc * 32 + hi * 16;
    int tap = kbase >> 7;              // kk = tap*128 + ci
    int cibase = kbase & 127;
    int q = pos + tap - 4;
    bool ok = pvalid && q >= 0 && q < L;
    int qc = iclamp(q, 0, L - 1);
    const float* src = h + (size_t)(img * C + cibase) * HW + sOfs + qc * qMul;
    v16h b;
#pragma unroll
    for (int j = 0; j < 16; ++j) {
      float xv = src[(size_t)j * HW];  // unconditional, clamped address
      b[j] = ok ? (_Float16)xv : (_Float16)0.f;
    }
    acc = __builtin_amdgcn_wmma_f32_16x16x32_f16(false, a, false, b,
                                                 (short)0, acc, false, false);
  }
  int dOfs = axisVert ? dstS * Wh : dstS;
#pragma unroll
  for (int r = 0; r < 8; ++r) {
    int co = coTile * 16 + r + hi * 8;
    if (pvalid) {
      size_t idx = (size_t)(img * C + co) * HW + dOfs + pos * qMul;
      float v = acc[r];
      h[idx] += (v > 0.f ? v : 0.f);
    }
  }
}

// ---------------------------------------------------------------------------
static const int STATS_B = 64;

template<int KH, int KW, int CI>
static void conv_vec(const _Float16* in, const _Float16* wf, const float* b,
                     float* out, int H, int W, int Co, int pad, int dil,
                     hipStream_t s) {
  const int NT = 4;
  int P = 4 * H * W;
  int coTiles = Co / 16;
  int grpTiles = (P + NT * 16 - 1) / (NT * 16);
  long waves = (long)coTiles * grpTiles;
  conv_wmma_vec_kernel<KH, KW, CI, NT><<<(unsigned)((waves + 7) / 8), 256, 0, s>>>(
      in, wf, b, out, H, W, Co, pad, dil, coTiles, grpTiles);
}

static void launch_repack(const float* w, _Float16* frag, int Co, int Ci, int khkw,
                          hipStream_t s) {
  int coTiles = (Co + 15) / 16;
  int kChunks = (Ci * khkw + 31) / 32;
  size_t tot = (size_t)coTiles * kChunks * 512;
  repack_w_kernel<<<(unsigned)((tot + 255) / 256), 256, 0, s>>>(
      w, frag, Co, Ci, khkw, coTiles, kChunks);
}

static void run_stats(const float* raw, float* part, float* st, int C,
                      size_t total, hipStream_t s) {
  stats1_kernel<<<STATS_B, 256, 0, s>>>(raw, part, C, total);
  stats2_kernel<<<1, 256, 0, s>>>(part, st, C, STATS_B);
}

static int ilog2(int v) { int r = 0; while ((1 << r) < v) ++r; return r; }

extern "C" void kernel_launch(void* const* d_in, const int* in_sizes, int n_in,
                              void* d_out, int out_size, void* d_ws, size_t ws_size,
                              hipStream_t stream) {
  (void)in_sizes; (void)n_in; (void)out_size; (void)ws_size;

  // Input order (setup_inputs insertion order):
  // 0: x ; 1+4i..: block i {w,b,g,bt} ; 49..54: l1_{w1,g1,b1,w2,g2,b2} ; 55..58: mp_{ud,du,lr,rl}
  const float* x = (const float*)d_in[0];
  float* outF = (float*)d_out;
  float* ws = (float*)d_ws;

  const size_t AMAX = 14745600;                  // 4*16*288*800 elements
  float*     raw  = ws;                          // NHWC f32 conv output [AMAX]
  _Float16*  actA = (_Float16*)(ws + AMAX);      // NHWC f16 act [AMAX halfs]
  _Float16*  actB = actA + AMAX;                 // NHWC f16 act [AMAX halfs]
  float*     hbuf = ws + 2 * AMAX;               // NCHW f32 h [460800]
  float*     st   = hbuf + 460800;               // [512]
  float*     part = st + 512;                    // [STATS_B * 512]
  _Float16*  wfrag  = (_Float16*)(part + (size_t)STATS_B * 512);  // <= 294912 halfs
  _Float16*  mpfrag = wfrag + 524288;            // 4 * 147456 halfs

  const int chans[12] = {16,16,16,32,32,32,64,64,64,128,128,128};
  const size_t featOff[3] = {0, 3686400, 5529600};

  // convert input image to NHWC f16
  {
    size_t nx = (size_t)4 * 3 * 288 * 800;
    cvt_in_kernel<<<(unsigned)((nx + 255) / 256), 256, 0, stream>>>(x, actA);
  }

  _Float16* bufs[2] = {actA, actB};
  int cb = 0, fi = 0;
  int H = 288, W = 800, Ci = 3;

  for (int i = 0; i < 12; ++i) {
    const float* wgt = (const float*)d_in[1 + 4 * i + 0];
    const float* bb  = (const float*)d_in[1 + 4 * i + 1];
    const float* gg  = (const float*)d_in[1 + 4 * i + 2];
    const float* bt  = (const float*)d_in[1 + 4 * i + 3];
    int Co = chans[i];
    launch_repack(wgt, wfrag, Co, Ci, 9, stream);
    if (Ci == 3) {
      const int NT = 4;
      int P = 4 * H * W;
      int coTiles = Co / 16, grpTiles = (P + 63) / 64;
      long waves = (long)coTiles * grpTiles;
      conv_wmma_gen_kernel<3, 3, 3, NT><<<(unsigned)((waves + 7) / 8), 256, 0, stream>>>(
          bufs[cb], wfrag, bb, raw, H, W, Co, 1, 1, coTiles, grpTiles);
    } else if (Ci == 16)  conv_vec<3, 3, 16 >(bufs[cb], wfrag, bb, raw, H, W, Co, 1, 1, stream);
    else if (Ci == 32)    conv_vec<3, 3, 32 >(bufs[cb], wfrag, bb, raw, H, W, Co, 1, 1, stream);
    else if (Ci == 64)    conv_vec<3, 3, 64 >(bufs[cb], wfrag, bb, raw, H, W, Co, 1, 1, stream);
    else                  conv_vec<3, 3, 128>(bufs[cb], wfrag, bb, raw, H, W, Co, 1, 1, stream);

    size_t total = (size_t)4 * H * W * Co;
    run_stats(raw, part, st, Co, total, stream);
    float cnt = (float)(4 * H * W);
    if ((i + 1) % 3 == 0) {
      float* feat = (fi < 3) ? (outF + featOff[fi]) : nullptr;
      size_t pt = (size_t)4 * Co * (H / 2) * (W / 2);
      bn_relu_pool_kernel<<<(unsigned)((pt + 255) / 256), 256, 0, stream>>>(
          raw, st, gg, bt, bufs[1 - cb], feat, Co, ilog2(Co), H, W, cnt);
      ++fi; H /= 2; W /= 2;
    } else {
      bn_relu_f16_kernel<<<(unsigned)((total + 255) / 256), 256, 0, stream>>>(
          raw, st, gg, bt, bufs[1 - cb], Co, total, cnt);
    }
    cb = 1 - cb;
    Ci = Co;
  }

  // h (4,18,50,128) NHWC f16 in bufs[cb]
  const float* l1w1 = (const float*)d_in[49];
  const float* l1g1 = (const float*)d_in[50];
  const float* l1b1 = (const float*)d_in[51];
  const float* l1w2 = (const float*)d_in[52];
  const float* l1g2 = (const float*)d_in[53];
  const float* l1b2 = (const float*)d_in[54];

  // dilated 3x3, 128->256, pad=4 dil=4
  launch_repack(l1w1, wfrag, 256, 128, 9, stream);
  conv_vec<3, 3, 128>(bufs[cb], wfrag, nullptr, raw, 18, 50, 256, 4, 4, stream);
  {
    size_t total = (size_t)4 * 900 * 256;
    run_stats(raw, part, st, 256, total, stream);
    bn_relu_f16_kernel<<<(unsigned)((total + 255) / 256), 256, 0, stream>>>(
        raw, st, l1g1, l1b1, bufs[1 - cb], 256, total, 3600.f);
  }
  cb = 1 - cb;

  // 1x1, 256->128 ; BN+ReLU -> NCHW f32 hbuf for SCNN
  launch_repack(l1w2, wfrag, 128, 256, 1, stream);
  conv_vec<1, 1, 256>(bufs[cb], wfrag, nullptr, raw, 18, 50, 128, 0, 1, stream);
  {
    size_t total = (size_t)4 * 900 * 128;
    run_stats(raw, part, st, 128, total, stream);
    bn_relu_nchw_kernel<<<(unsigned)((total + 255) / 256), 256, 0, stream>>>(
        raw, st, l1g2, l1b2, hbuf, 128, 7, 18, 50, total, 3600.f);
  }

  // SCNN message passing: repack 4 kernel sets once, then sequential steps
  const size_t MPFRAG = 147456;
  for (int s4 = 0; s4 < 4; ++s4)
    launch_repack((const float*)d_in[55 + s4], mpfrag + (size_t)s4 * MPFRAG,
                  128, 128, 9, stream);

  auto mp = [&](int set, int axisVert, int src, int dst) {
    int L = axisVert ? 50 : 18;
    int pixT = (4 * L + 15) / 16;
    unsigned grid = (unsigned)((8 * pixT + 7) / 8);
    mp_step_wmma_kernel<<<grid, 256, 0, stream>>>(
        hbuf, mpfrag + (size_t)set * MPFRAG, axisVert, src, dst, pixT);
  };
  for (int sI = 1;  sI < 18;  ++sI) mp(0, 1, sI - 1, sI);
  for (int sI = 16; sI >= 0;  --sI) mp(1, 1, sI + 1, sI);
  for (int sI = 1;  sI < 50;  ++sI) mp(2, 0, sI - 1, sI);
  for (int sI = 48; sI >= 0;  --sI) mp(3, 0, sI + 1, sI);

  hipMemcpyAsync(outF + 6451200, hbuf, (size_t)460800 * sizeof(float),
                 hipMemcpyDeviceToDevice, stream);
}
